// GRNN_46840913330241
// MI455X (gfx1250) — compile-verified, matrix operand order checked by reading, not run on hardware
//
#include <hip/hip_runtime.h>
#include <hip/hip_bf16.h>

// GRNN: out = rownorm(exp(-pairdist2(x)/2)) @ x @ W^T + b
// N=8192, D=512, OUT=512. Flash-style fusion, bf16 WMMA for the two N^2*D
// GEMMs (137 GFLOP, compute-bound: AI ~2000 FLOP/B), f32 WMMA 16x16x4 for the
// final projection. x^T panels double-buffered in LDS via async-to-LDS DMA.

typedef __attribute__((ext_vector_type(16))) __bf16 v16bf;
typedef __attribute__((ext_vector_type(8)))  __bf16 v8bf;
typedef __attribute__((ext_vector_type(8)))  float  v8f;
typedef __attribute__((ext_vector_type(4)))  float  v4f;
typedef __attribute__((ext_vector_type(2)))  float  v2f;
typedef __attribute__((ext_vector_type(4)))  int    v4i;

#define NROWS 8192
#define DDIM  512
#define NOUT  512
#define LB_STRIDE 80            // 64B row of x^T panel + 16B pad (16B aligned, conflict-free)
#define LB_BUF (DDIM * LB_STRIDE)  // 40960 B per buffer

union FragBF { v16bf f; v8bf h[2]; __bf16 e[16]; };

// CDNA5 async copy: global -> LDS, tracked by ASYNCcnt (no VGPR data path).
__device__ __forceinline__ void async_ld_b128(unsigned lds_addr, const void* gptr) {
  asm volatile("global_load_async_to_lds_b128 %0, %1, off"
               :: "v"(lds_addr), "v"((unsigned long long)(uintptr_t)gptr)
               : "memory");
}
__device__ __forceinline__ void wait_async0() {
  asm volatile("s_wait_asynccnt 0x0" ::: "memory");
}

// ---------------- Kernel 1: sq_norms + bf16 copies (row-major and transposed)
__global__ void __launch_bounds__(256) grnn_prep(const float* __restrict__ x,
                                                 float* __restrict__ sqn,
                                                 __bf16* __restrict__ xh,
                                                 __bf16* __restrict__ xhT) {
  const int lane = threadIdx.x & 31;
  const int wave = threadIdx.x >> 5;
  const int row  = blockIdx.x * 8 + wave;
  const float* xr = x + (size_t)row * DDIM + lane * 16;
  float s = 0.f;
  union { __bf16 e[16]; v8bf v[2]; } hv;
#pragma unroll
  for (int i = 0; i < 4; ++i) {
    v4f v = *(const v4f*)(xr + i * 4);
#pragma unroll
    for (int j = 0; j < 4; ++j) { s += v[j] * v[j]; hv.e[i * 4 + j] = (__bf16)v[j]; }
  }
  v8bf* dst = (v8bf*)(xh + (size_t)row * DDIM + lane * 16);
  dst[0] = hv.v[0];
  dst[1] = hv.v[1];
#pragma unroll
  for (int i = 0; i < 16; ++i)
    xhT[(size_t)(lane * 16 + i) * NROWS + row] = hv.e[i];
#pragma unroll
  for (int off = 16; off > 0; off >>= 1) s += __shfl_xor(s, off, 32);
  if (lane == 0) sqn[row] = s;
}

// ---------------- Kernel 2: fused flash smoother: outs = rownorm(exp(S - n/2)) @ x
// One wave owns 16 rows (full-D f32 accumulator). 4 waves/WG share a
// double-buffered, async-staged 32-column panel of x^T for GEMM2's B frags.
__global__ void __launch_bounds__(128) grnn_smooth(const __bf16* __restrict__ xh,
                                                   const __bf16* __restrict__ xhT,
                                                   const float* __restrict__ sqn,
                                                   float* __restrict__ outs) {
  __shared__ __align__(16) char lB[2 * LB_BUF];  // 80KB of the 320KB WGP pool
  const int tid  = threadIdx.x;
  const int lane = tid & 31;
  const int wave = tid >> 5;
  const int l15  = lane & 15;
  const int hb   = lane >> 4;
  const int i0   = (blockIdx.x * 4 + wave) * 16;
  const unsigned lb_base = (unsigned)(uintptr_t)&lB[0];

  // Preload GEMM1 B-fragments: our 16 rows of x (bf16), all K. 128 VGPRs.
  v16bf xr[16];
  {
    const char* gR = (const char*)(xh + (size_t)i0 * DDIM);
#pragma unroll
    for (int kc8 = 0; kc8 < 16; ++kc8) {
      const char* p = gR + l15 * (DDIM * 2) + (kc8 * 32 + hb * 8) * 2;
      FragBF u;
      u.h[0] = *(const v8bf*)(p);
      u.h[1] = *(const v8bf*)(p + 32);
      xr[kc8] = u.f;
    }
  }
  const float sqm = sqn[i0 + l15];  // row norm for m = lane%16

  v8f acc[32] = {};  // 16 rows x 512 cols f32
  float wl = 0.f;    // per-lane partial row sum (lane <-> row m)

  // Kick off async staging of panel 0.
  {
    const char* gB = (const char*)xhT;
#pragma unroll
    for (int it = 0; it < 16; ++it) {
      int c   = it * 128 + tid;
      int row = c >> 2;
      int off = (c & 3) << 4;
      async_ld_b128(lb_base + row * LB_STRIDE + off,
                    gB + (size_t)row * (NROWS * 2) + off);
    }
  }

  for (int jb = 0; jb < NROWS / 32; ++jb) {
    const int j0 = jb * 32;
    wait_async0();      // our wave's async fills done
    __syncthreads();    // everyone's fills done; prior buffer free for reuse

    // Stage next panel asynchronously into the other buffer while we compute.
    if (jb + 1 < NROWS / 32) {
      const char* gB = (const char*)xhT + (size_t)(j0 + 32) * 2;
      const unsigned nb = lb_base + ((jb + 1) & 1) * LB_BUF;
#pragma unroll
      for (int it = 0; it < 16; ++it) {
        int c   = it * 128 + tid;
        int row = c >> 2;
        int off = (c & 3) << 4;
        async_ld_b128(nb + row * LB_STRIDE + off,
                      gB + (size_t)row * (NROWS * 2) + off);
      }
    }
    const char* cur = &lB[(jb & 1) * LB_BUF];

    // GEMM1: S^T (2 tiles of 16x16) = X_cols . X_rows^T over K=512.
    v8f c0 = {}, c1 = {};
    const char* gC = (const char*)(xh + (size_t)j0 * DDIM);
    // Warm WGP$ one iteration ahead for the next A panel.
    if (j0 + 32 < NROWS)
      __builtin_prefetch((const char*)(xh + (size_t)(j0 + 32 + l15) * DDIM) + hb * 512, 0, 3);
#pragma unroll
    for (int kc8 = 0; kc8 < 16; ++kc8) {
      const int ko = (kc8 * 32 + hb * 8) * 2;
      FragBF a0, a1;
      const char* p0 = gC + l15 * (DDIM * 2) + ko;
      const char* p1 = gC + (16 + l15) * (DDIM * 2) + ko;
      a0.h[0] = *(const v8bf*)(p0); a0.h[1] = *(const v8bf*)(p0 + 32);
      a1.h[0] = *(const v8bf*)(p1); a1.h[1] = *(const v8bf*)(p1 + 32);
      c0 = __builtin_amdgcn_wmma_f32_16x16x32_bf16(false, a0.f, false, xr[kc8],
                                                   (short)0, c0, false, false);
      c1 = __builtin_amdgcn_wmma_f32_16x16x32_bf16(false, a1.f, false, xr[kc8],
                                                   (short)0, c1, false, false);
    }

    // RBF weights: w = exp(min(0, S - (|xi|^2+|xj|^2)/2)). S^T C-layout packs
    // per-lane straight into the bf16 A fragment of the second GEMM.
    const v4f* pj0 = (const v4f*)(sqn + j0 + hb * 8);
    const v4f* pj1 = (const v4f*)(sqn + j0 + 16 + hb * 8);
    float sj0[8], sj1[8];
    {
      v4f t0 = pj0[0], t1 = pj0[1], t2 = pj1[0], t3 = pj1[1];
#pragma unroll
      for (int v = 0; v < 4; ++v) {
        sj0[v] = t0[v]; sj0[v + 4] = t1[v];
        sj1[v] = t2[v]; sj1[v + 4] = t3[v];
      }
    }
    FragBF aw;
#pragma unroll
    for (int v = 0; v < 8; ++v) {
      float e0 = __expf(fminf(0.f, c0[v] - 0.5f * (sqm + sj0[v])));
      float e1 = __expf(fminf(0.f, c1[v] - 0.5f * (sqm + sj1[v])));
      wl += e0 + e1;
      aw.e[v]     = (__bf16)e0;
      aw.e[8 + v] = (__bf16)e1;
    }

    // GEMM2: acc(16x512) += w(16x32) @ X_cols(32x512); B frags from LDS x^T.
#pragma unroll
    for (int nt = 0; nt < 32; ++nt) {
      FragBF bB;
      const char* bb = cur + (nt * 16 + l15) * LB_STRIDE + hb * 16;
      bB.h[0] = *(const v8bf*)(bb);
      bB.h[1] = *(const v8bf*)(bb + 32);
      acc[nt] = __builtin_amdgcn_wmma_f32_16x16x32_bf16(false, aw.f, false, bB.f,
                                                        (short)0, acc[nt], false, false);
    }
  }

  // Row-normalize and store smoothed x (f32).
  float wsum = wl + __shfl_xor(wl, 16, 32);
  float winv = 1.f / wsum;
  float rv[8];
#pragma unroll
  for (int r = 0; r < 8; ++r) rv[r] = __shfl(winv, r + 8 * hb, 32);
#pragma unroll
  for (int nt = 0; nt < 32; ++nt)
#pragma unroll
    for (int r = 0; r < 8; ++r)
      outs[(size_t)(i0 + r + 8 * hb) * DDIM + nt * 16 + l15] = acc[nt][r] * rv[r];
}

// ---------------- Kernel 3: out = outs @ W^T + b, full f32 via WMMA 16x16x4.
__global__ void __launch_bounds__(32) grnn_linear(const float* __restrict__ outs,
                                                  const float* __restrict__ W,
                                                  const float* __restrict__ b,
                                                  float* __restrict__ out) {
  const int lane = threadIdx.x & 31;
  const int l15  = lane & 15;
  const int h    = lane >> 4;
  const int i0   = blockIdx.x * 16;
  const int n0   = blockIdx.y * 128;
  v8f acc[8];
#pragma unroll
  for (int nt = 0; nt < 8; ++nt) {
    float bv = b[n0 + nt * 16 + l15];
#pragma unroll
    for (int r = 0; r < 8; ++r) acc[nt][r] = bv;
  }
  const float* arow = outs + (size_t)(i0 + l15) * DDIM;
  for (int k0 = 0; k0 < DDIM; k0 += 4) {
    v2f a = *(const v2f*)(arow + k0 + 2 * h);
#pragma unroll
    for (int nt = 0; nt < 8; ++nt) {
      v2f bb = *(const v2f*)(W + (size_t)(n0 + nt * 16 + l15) * DDIM + k0 + 2 * h);
      acc[nt] = __builtin_amdgcn_wmma_f32_16x16x4_f32(false, a, false, bb,
                                                      (short)0, acc[nt], false, false);
    }
  }
#pragma unroll
  for (int nt = 0; nt < 8; ++nt)
#pragma unroll
    for (int r = 0; r < 8; ++r)
      out[(size_t)(i0 + r + 8 * h) * NOUT + n0 + nt * 16 + l15] = acc[nt][r];
}

extern "C" void kernel_launch(void* const* d_in, const int* in_sizes, int n_in,
                              void* d_out, int out_size, void* d_ws, size_t ws_size,
                              hipStream_t stream) {
  (void)in_sizes; (void)n_in; (void)out_size; (void)ws_size;
  const float* x = (const float*)d_in[0];
  const float* W = (const float*)d_in[1];
  const float* b = (const float*)d_in[2];
  float* out = (float*)d_out;

  char* ws = (char*)d_ws;
  float*  sqn  = (float*)(ws);                                      // 32 KB
  __bf16* xh   = (__bf16*)(ws + 32768);                             // 8 MB
  __bf16* xhT  = (__bf16*)(ws + 32768 + (size_t)8 * 1024 * 1024);   // 8 MB
  float*  outs = (float*)(ws + 32768 + (size_t)16 * 1024 * 1024);   // 16 MB

  grnn_prep<<<NROWS / 8, 256, 0, stream>>>(x, sqn, xh, xhT);
  grnn_smooth<<<NROWS / 64, 128, 0, stream>>>(xh, xhT, sqn, outs);
  grnn_linear<<<dim3(NROWS / 16, NOUT / 128), 32, 0, stream>>>(outs, W, b, out);
}
